// GraphDiscriminator_37177236914939
// MI455X (gfx1250) — matrix operand order repaired and checked by reading, not compiled, not run on hardware
//
#include <hip/hip_runtime.h>

#define NN 50000
#define NE 800000
#define NG 512
#define INDIM 128
#define HID 64

typedef __attribute__((ext_vector_type(16))) __bf16 v16bf;
typedef __attribute__((ext_vector_type(8)))  float  v8f;

// ---------------------------------------------------------------------------
// Zero fill (float4 granularity; all buffers are multiples of 16 bytes)
// ---------------------------------------------------------------------------
__global__ void zero_kernel(float4* __restrict__ p, long long n4) {
    long long i = (long long)blockIdx.x * blockDim.x + threadIdx.x;
    long long stride = (long long)gridDim.x * blockDim.x;
    float4 z = {0.f, 0.f, 0.f, 0.f};
    for (; i < n4; i += stride) p[i] = z;
}

// ---------------------------------------------------------------------------
// Convert a [K,64] fp32 weight into WMMA B-fragment-ordered bf16:
// out[((kt*4 + nt)*32 + lane)*16 + i] = W[(kt*32 + (lane>>4)*16 + i)*64 + nt*16 + (lane&15)]
// so each lane's 16-half B fragment is one contiguous 32B load in the GEMM.
// ---------------------------------------------------------------------------
__global__ void wfrag_kernel(const float* __restrict__ W, __bf16* __restrict__ out, int KDIM) {
    int idx = blockIdx.x * blockDim.x + threadIdx.x;
    int total = KDIM * 64;
    if (idx >= total) return;
    int i    = idx & 15;
    int lane = (idx >> 4) & 31;
    int nt   = (idx >> 9) & 3;
    int kt   = idx >> 11;
    int k = kt * 32 + (lane >> 4) * 16 + i;
    int n = nt * 16 + (lane & 15);
    out[idx] = (__bf16)W[k * 64 + n];
}

// ---------------------------------------------------------------------------
// Edge scatter: agg[dst[e]] += X[src[e]]  (fp32 atomics; L2-resident)
// K is a compile-time power of two -> pure shift/mask indexing, 32-bit math.
// ---------------------------------------------------------------------------
template <int K>
__global__ void scatter_kernel(const float* __restrict__ X,
                               const long long* __restrict__ src,
                               const long long* __restrict__ dst,
                               float* __restrict__ agg) {
    constexpr int CH = K >> 2;              // float4 chunks per edge
    constexpr int total = NE * CH;          // 25.6M / 12.8M, fits in int
    int id = blockIdx.x * blockDim.x + threadIdx.x;
    const int stride = gridDim.x * blockDim.x;
    for (; id < total; id += stride) {
        int e = id / CH;                    // constant power of two -> shift
        int c = (id - e * CH) * 4;
        int s = (int)src[e];
        int d = (int)dst[e];
        float4 v = *(const float4*)(X + (size_t)s * K + c);
        float* ap = agg + (size_t)d * K + c;
        atomicAdd(ap + 0, v.x);
        atomicAdd(ap + 1, v.y);
        atomicAdd(ap + 2, v.z);
        atomicAdd(ap + 3, v.w);
    }
}

// ---------------------------------------------------------------------------
// A-fragment loader: 16x32 bf16 tile slice for one lane, converted from fp32.
// Lane layout: lanes g*16+row; halves 0..15 -> K=k0+16g+i (matches wfrag B order).
// One contiguous 64B fp32 run per lane.
// ---------------------------------------------------------------------------
__device__ __forceinline__ v16bf load_a_frag(const float* __restrict__ xrow, int k0, int g) {
    const float* p = xrow + k0 + g * 16;
    const float* q = xrow + k0 + g * 16 + 8;
    v16bf a;
#pragma unroll
    for (int i = 0; i < 8; ++i) a[i] = (__bf16)p[i];
#pragma unroll
    for (int i = 0; i < 8; ++i) a[8 + i] = (__bf16)q[i];
    return a;
}

// ---------------------------------------------------------------------------
// Fused dual-GEMM + ReLU: H = relu(Xself @ W1 + Xagg @ W2), output [NN,64].
// Block = 128 threads = 4 waves. Each WAVE owns one 16-row m-tile and keeps
// 4 accumulators (all 4 n-tiles of the 64-wide output), so each A fragment
// is loaded + converted ONCE and fed to 4 v_wmma_f32_16x16x32_bf16 per
// K-step (4x less activation traffic than one-tile-per-wave). Waves whose
// m-tile is out of range exit uniformly before any WMMA (EXEC stays all-1s).
// ---------------------------------------------------------------------------
template <int KDIM>
__global__ __launch_bounds__(128) void gemm_relu_kernel(
    const float* __restrict__ Xself, const float* __restrict__ Xagg,
    const __bf16* __restrict__ W1f, const __bf16* __restrict__ W2f,
    float* __restrict__ H) {
    const int wave = threadIdx.x >> 5;
    const int mt   = blockIdx.x * 4 + wave;      // m-tile index
    if (mt >= NN / 16) return;                   // uniform per-wave exit
    const int m0   = mt * 16;
    const int lane = threadIdx.x & 31;
    const int g    = lane >> 4;                  // lane group
    const int r16  = lane & 15;                  // A row / C column index

    const float* xs = Xself + (size_t)(m0 + r16) * KDIM;
    const float* xa = Xagg  + (size_t)(m0 + r16) * KDIM;

    v8f acc[4];
#pragma unroll
    for (int nt = 0; nt < 4; ++nt) acc[nt] = (v8f){0.f, 0.f, 0.f, 0.f, 0.f, 0.f, 0.f, 0.f};

#pragma unroll
    for (int kt = 0; kt < KDIM / 32; ++kt) {
        const int k0 = kt * 32;
        v16bf a1 = load_a_frag(xs, k0, g);
        v16bf a2 = load_a_frag(xa, k0, g);
#pragma unroll
        for (int nt = 0; nt < 4; ++nt) {
            const size_t fb = (((size_t)kt * 4 + nt) * 32 + lane) * 16;
            v16bf b1 = *(const v16bf*)(W1f + fb);
            acc[nt] = __builtin_amdgcn_wmma_f32_16x16x32_bf16(
                false, a1, false, b1, (short)0, acc[nt], false, false);
            v16bf b2 = *(const v16bf*)(W2f + fb);
            acc[nt] = __builtin_amdgcn_wmma_f32_16x16x32_bf16(
                false, a2, false, b2, (short)0, acc[nt], false, false);
        }
    }
#pragma unroll
    for (int nt = 0; nt < 4; ++nt) {
#pragma unroll
        for (int r = 0; r < 8; ++r) {
            float v = acc[nt][r];
            v = v > 0.f ? v : 0.f;
            // C/D layout: M = r + 8*g, N = lane&15
            H[(size_t)(m0 + r + 8 * g) * HID + nt * 16 + r16] = v;
        }
    }
}

// ---------------------------------------------------------------------------
// Segment-sum pooling: sums[batch[n]] += h[n]; counts[batch[n]] += 1
// ---------------------------------------------------------------------------
__global__ void pool_kernel(const float* __restrict__ h,
                            const long long* __restrict__ batch,
                            float* __restrict__ sums, float* __restrict__ counts) {
    int idx = blockIdx.x * blockDim.x + threadIdx.x;
    if (idx >= NN * HID) return;
    int n = idx >> 6;
    int c = idx & 63;
    int b = (int)batch[n];
    atomicAdd(&sums[(size_t)b * HID + c], h[idx]);
    if (c == 0) atomicAdd(&counts[b], 1.0f);
}

// ---------------------------------------------------------------------------
// Classifier: out[g] = relu(pooled @ cW1 + cb1) @ cW2 + cb2  (tiny; VALU)
// ---------------------------------------------------------------------------
__global__ __launch_bounds__(64) void cls_kernel(
    const float* __restrict__ sums, const float* __restrict__ counts,
    const float* __restrict__ cW1, const float* __restrict__ cb1,
    const float* __restrict__ cW2, const float* __restrict__ cb2,
    float* __restrict__ out) {
    __shared__ float pooled[HID];
    __shared__ float hred[HID];
    const int gidx = blockIdx.x;
    const int j = threadIdx.x;
    float cnt = counts[gidx];
    cnt = cnt < 1.f ? 1.f : cnt;
    pooled[j] = sums[(size_t)gidx * HID + j] / cnt;
    __syncthreads();
    float acc = cb1[j];
#pragma unroll 8
    for (int k = 0; k < HID; ++k) acc += pooled[k] * cW1[k * HID + j];
    acc = acc > 0.f ? acc : 0.f;
    hred[j] = acc * cW2[j];
    __syncthreads();
    for (int off = 32; off > 0; off >>= 1) {
        if (j < off) hred[j] += hred[j + off];
        __syncthreads();
    }
    if (j == 0) out[gidx] = hred[0] + cb2[0];
}

// ---------------------------------------------------------------------------
extern "C" void kernel_launch(void* const* d_in, const int* in_sizes, int n_in,
                              void* d_out, int out_size, void* d_ws, size_t ws_size,
                              hipStream_t stream) {
    const float*     x     = (const float*)d_in[0];
    const long long* eidx  = (const long long*)d_in[1];
    const long long* batch = (const long long*)d_in[2];
    const float* W1_1 = (const float*)d_in[3];
    const float* W2_1 = (const float*)d_in[4];
    const float* W1_2 = (const float*)d_in[5];
    const float* W2_2 = (const float*)d_in[6];
    const float* W1_3 = (const float*)d_in[7];
    const float* W2_3 = (const float*)d_in[8];
    const float* cW1  = (const float*)d_in[9];
    const float* cb1  = (const float*)d_in[10];
    const float* cW2  = (const float*)d_in[11];
    const float* cb2  = (const float*)d_in[12];
    float* out = (float*)d_out;

    const long long* src = eidx;            // edge_index[0]
    const long long* dst = eidx + NE;       // edge_index[1]

    // Workspace layout (all offsets 256B aligned)
    char* ws = (char*)d_ws;
    float* agg   = (float*)(ws);              // 50000*128*4 = 25,600,000 B
    float* hA    = (float*)(ws + 25600000);   // 50000*64*4  = 12,800,000 B
    float* hB    = (float*)(ws + 38400000);   // 12,800,000 B
    float* pool  = (float*)(ws + 51200000);   // 512*64*4 = 131,072 B
    float* cnts  = (float*)(ws + 51331072);   // 2,048 B
    __bf16* wf   = (__bf16*)(ws + 51333120);  // 65,536 B of bf16 weight frags
    __bf16* w11f = wf;
    __bf16* w21f = wf + 8192;
    __bf16* w12f = wf + 16384;
    __bf16* w22f = wf + 20480;
    __bf16* w13f = wf + 24576;
    __bf16* w23f = wf + 28672;

    // --- weight fragment conversion (fp32 -> bf16, WMMA B order) ---
    wfrag_kernel<<<32, 256, 0, stream>>>(W1_1, w11f, INDIM);
    wfrag_kernel<<<32, 256, 0, stream>>>(W2_1, w21f, INDIM);
    wfrag_kernel<<<16, 256, 0, stream>>>(W1_2, w12f, HID);
    wfrag_kernel<<<16, 256, 0, stream>>>(W2_2, w22f, HID);
    wfrag_kernel<<<16, 256, 0, stream>>>(W1_3, w13f, HID);
    wfrag_kernel<<<16, 256, 0, stream>>>(W2_3, w23f, HID);

    const int GEMM_GRID = (NN / 16 + 3) / 4;  // 782 blocks x 4 waves (m-tiles)

    // --- layer 1 (K=128) ---
    zero_kernel<<<2048, 256, 0, stream>>>((float4*)agg, (long long)NN * INDIM / 4);
    scatter_kernel<INDIM><<<4096, 256, 0, stream>>>(x, src, dst, agg);
    gemm_relu_kernel<INDIM><<<GEMM_GRID, 128, 0, stream>>>(x, agg, w11f, w21f, hA);

    // --- layer 2 (K=64) ---
    zero_kernel<<<2048, 256, 0, stream>>>((float4*)agg, (long long)NN * HID / 4);
    scatter_kernel<HID><<<4096, 256, 0, stream>>>(hA, src, dst, agg);
    gemm_relu_kernel<HID><<<GEMM_GRID, 128, 0, stream>>>(hA, agg, w12f, w22f, hB);

    // --- layer 3 (K=64) ---
    zero_kernel<<<2048, 256, 0, stream>>>((float4*)agg, (long long)NN * HID / 4);
    scatter_kernel<HID><<<4096, 256, 0, stream>>>(hB, src, dst, agg);
    gemm_relu_kernel<HID><<<GEMM_GRID, 128, 0, stream>>>(hB, agg, w13f, w23f, hA);

    // --- pooling + classifier ---
    zero_kernel<<<64, 256, 0, stream>>>((float4*)pool, (long long)NG * HID / 4);
    zero_kernel<<<1, 128, 0, stream>>>((float4*)cnts, (long long)NG / 4);
    pool_kernel<<<(NN * HID + 255) / 256, 256, 0, stream>>>(hA, batch, pool, cnts);
    cls_kernel<<<NG, 64, 0, stream>>>(pool, cnts, cW1, cb1, cW2, cb2, out);
}
